// OutProductMean_37572373905864
// MI455X (gfx1250) — matrix operand order repaired and checked by reading, not compiled, not run on hardware
//
#include <hip/hip_runtime.h>
#include <hip/hip_bf16.h>

// ---------------------------------------------------------------------------
// OuterProductMean for MI455X (gfx1250, wave32, WMMA, async-LDS staging).
//
//   Z[i,j,o] = ( sum_k Wo[o,k] * G_i[k,j] + bo[o] ) / (0.001 + norm[i,j]) + Zraw
//   G_i[(c,d), j] = sum_s L[s,i,c] * R[s,j,d]
//
// prep_kernel : LayerNorm (VALU) + [Wa;Wb] projection as WMMA GEMM
//               (M=16 rows, N=64 proj, K=64 feat per tile), mask+bias epilogue,
//               LDS-transposed coalesced bf16 writes of Lt[(i*32+c)][s],
//               Rt[(j*32+d)][s]  (s contiguous -> WMMA frags = ds_load_b128).
// wo_convert  : Wo fp32 -> bf16 row-major [o][k], k = c*32+d.
// opm_main    : per workgroup = (one i, 32 j's), 8 waves.
//   Stage A: G = L_i^T x R   (M=32, N=1024, K=256) wmma_f32_16x16x32_bf16,
//            inputs staged with global_load_async_to_lds_b128 (ASYNCcnt).
//   Stage B: Z = Wo   x G    (M=128, N=32, K=1024) wmma_f32_16x16x32_bf16.
//   G never touches HBM (604 MB avoided); lives in LDS in two 512-K halves.
//
// Roofline: ~116 GFLOP total, ~180 MB HBM min traffic -> memory bound
// (~8 us at 23.3 TB/s); bf16 WMMA makes compute negligible.
// Workspace: Lt 6MB + Rt 6MB + WoB 256KB = 12.6 MB (assumed <= ws_size).
// ---------------------------------------------------------------------------

#define S_DIM 256
#define R_DIM 384
#define F_DIM 64
#define P_DIM 32
#define O_DIM 128
#define K2    1024           // P_DIM*P_DIM
#define JB    32             // j's per workgroup

typedef __attribute__((ext_vector_type(16))) __bf16 v16bf;
typedef __attribute__((ext_vector_type(8)))  float  v8f;

union FragB16 { uint4 q[2]; v16bf v; };
union PK8     { __bf16 h[8]; uint4 q; };

// ---- CDNA5 async global->LDS copy (ISA 15.18.3 opc 96..98, ASYNCcnt) ------
__device__ __forceinline__ uint32_t lds_off32(const void* p) {
  // flat aperture maps addr[31:0] to the wave's LDS offset (ISA 10.2)
  return (uint32_t)(uintptr_t)p;
}
__device__ __forceinline__ void async_b128(uint32_t lds, uint32_t goff,
                                           const void* base) {
  asm volatile("global_load_async_to_lds_b128 %0, %1, %2"
               :: "v"(lds), "v"(goff), "s"(base) : "memory");
}
__device__ __forceinline__ void async_b64(uint32_t lds, uint32_t goff,
                                          const void* base) {
  asm volatile("global_load_async_to_lds_b64 %0, %1, %2"
               :: "v"(lds), "v"(goff), "s"(base) : "memory");
}
__device__ __forceinline__ void wait_async0() {
  asm volatile("s_wait_asynccnt 0x0" ::: "memory");
}

// =========================================================== prep (WMMA LN+proj)
// grid (24 r-tiles, 16 s-tiles), 256 threads, tile = 16 s x 16 r.
__global__ __launch_bounds__(256) void prep_kernel(
    const float* __restrict__ M, const float* __restrict__ mask,
    const float* __restrict__ gamma, const float* __restrict__ beta,
    const float* __restrict__ Wa, const float* __restrict__ ba,
    const float* __restrict__ Wb, const float* __restrict__ bb,
    __bf16* __restrict__ Lt, __bf16* __restrict__ Rt) {
  // LDS map (47360 B):
  //   A   [256 rows][72 bf16]  @0      (36864)  row = si*16+ri, padded 144B
  //   W   [64 rows][72 bf16]   @36864  (9216)   rows = [Wa;Wb]
  //   mk  [256 f32]            @46080  (1024)
  //   bias[64 f32]             @47104  (256)
  //   T   [64 n][16 r][16 s]   @0      (32768)  reuses A after WMMA
  __shared__ __align__(16) char pm[47360];
  constexpr int PA = 0, PW = 36864, PMK = 46080, PB = 47104, PT = 0;

  const int tid = threadIdx.x;
  const int wave = tid >> 5, lane = tid & 31;
  const int half = lane >> 4, ln16 = lane & 15;
  const int r0 = blockIdx.x * 16, s0 = blockIdx.y * 16;

  // stage combined weight matrix W[n=64][f=64] -> bf16
  {
    int n = tid >> 2, f0 = (tid & 3) * 16;
    const float* w = (n < P_DIM ? Wa + n * F_DIM : Wb + (n - P_DIM) * F_DIM) + f0;
    PK8 p0, p1;
#pragma unroll
    for (int u = 0; u < 8; ++u) { p0.h[u] = (__bf16)w[u]; p1.h[u] = (__bf16)w[u + 8]; }
    *(uint4*)(pm + PW + n * 144 + f0 * 2) = p0.q;
    *(uint4*)(pm + PW + n * 144 + f0 * 2 + 16) = p1.q;
  }
  if (tid < 64)
    ((float*)(pm + PB))[tid] = (tid < P_DIM ? ba[tid] : bb[tid - P_DIM]);
  ((float*)(pm + PMK))[tid] =
      mask[(size_t)(s0 + (tid >> 4)) * R_DIM + r0 + (tid & 15)];

  // LayerNorm of own row -> bf16 A tile
  {
    const float* mrow =
        M + ((size_t)(s0 + (tid >> 4)) * R_DIM + (r0 + (tid & 15))) * F_DIM;
    float v[F_DIM];
    float mu = 0.f;
#pragma unroll
    for (int f = 0; f < F_DIM; ++f) { v[f] = mrow[f]; mu += v[f]; }
    mu *= (1.f / F_DIM);
    float var = 0.f;
#pragma unroll
    for (int f = 0; f < F_DIM; ++f) { float d = v[f] - mu; var += d * d; }
    float rs = rsqrtf(var * (1.f / F_DIM) + 1e-5f);
#pragma unroll
    for (int qq = 0; qq < 8; ++qq) {
      PK8 pk;
#pragma unroll
      for (int u = 0; u < 8; ++u) {
        int f = qq * 8 + u;
        pk.h[u] = (__bf16)((v[f] - mu) * rs * gamma[f] + beta[f]);
      }
      *(uint4*)(pm + PA + tid * 144 + qq * 16) = pk.q;
    }
  }
  __syncthreads();

  // GEMM: per wave 2 tiles (si = wave*2+st), A[16r x 64f] x B[64f x 64n]
  FragB16 bfr[4][2];
#pragma unroll
  for (int nt = 0; nt < 4; ++nt)
#pragma unroll
    for (int kc = 0; kc < 2; ++kc) {
      const char* p = pm + PW + (nt * 16 + ln16) * 144 + (kc * 32 + 16 * half) * 2;
      bfr[nt][kc].q[0] = *(const uint4*)p;
      bfr[nt][kc].q[1] = *(const uint4*)(p + 16);
    }
  v8f acc[2][4];
#pragma unroll
  for (int a = 0; a < 2; ++a)
#pragma unroll
    for (int b = 0; b < 4; ++b)
      acc[a][b] = (v8f){0.f, 0.f, 0.f, 0.f, 0.f, 0.f, 0.f, 0.f};

#pragma unroll
  for (int st = 0; st < 2; ++st) {
    int si = wave * 2 + st;
    FragB16 afr[2];
#pragma unroll
    for (int kc = 0; kc < 2; ++kc) {
      const char* p = pm + PA + (si * 16 + ln16) * 144 + (kc * 32 + 16 * half) * 2;
      afr[kc].q[0] = *(const uint4*)p;
      afr[kc].q[1] = *(const uint4*)(p + 16);
    }
#pragma unroll
    for (int nt = 0; nt < 4; ++nt)
#pragma unroll
      for (int kc = 0; kc < 2; ++kc)
        acc[st][nt] = __builtin_amdgcn_wmma_f32_16x16x32_bf16(
            false, afr[kc].v, false, bfr[nt][kc].v, (short)0, acc[st][nt],
            false, false);
  }
  __syncthreads();

  // bias + mask, transpose into T[n][r][s]
  const float* biasL = (const float*)(pm + PB);
  const float* mkL   = (const float*)(pm + PMK);
#pragma unroll
  for (int st = 0; st < 2; ++st) {
    int si = wave * 2 + st;
#pragma unroll
    for (int nt = 0; nt < 4; ++nt) {
      int n = nt * 16 + ln16;
      float bsn = biasL[n];
#pragma unroll
      for (int vr = 0; vr < 8; ++vr) {
        int rl = vr + 8 * half;
        float val = (acc[st][nt][vr] + bsn) * mkL[si * 16 + rl];
        *(__bf16*)(pm + PT + ((n * 16 + rl) * 16 + si) * 2) = (__bf16)val;
      }
    }
  }
  __syncthreads();

  // coalesced bf16 output: each (n, r) is a contiguous 16-s run (32 B)
#pragma unroll
  for (int q = 0; q < 4; ++q) {
    int idx = q * 256 + tid;
    int n = idx >> 4, rl = idx & 15;
    const uint4* src = (const uint4*)(pm + PT + (n * 16 + rl) * 32);
    __bf16* dst = (n < P_DIM ? Lt : Rt) +
                  ((size_t)((r0 + rl) * P_DIM + (n & (P_DIM - 1)))) * S_DIM + s0;
    ((uint4*)dst)[0] = src[0];
    ((uint4*)dst)[1] = src[1];
  }
}

// ---------------------------------------------------------------- Wo convert
__global__ __launch_bounds__(256) void wo_convert(const float* __restrict__ Wo,
                                                  __bf16* __restrict__ WoB) {
  int t = blockIdx.x * 256 + threadIdx.x;
  if (t < O_DIM * K2) WoB[t] = (__bf16)Wo[t];
}

// ===================================================================== main
// LDS byte offsets (dynamic shared, total 50304 B)
#define A_OFF   0        // 32 rows * 80 B          (phase 1)
#define R_OFF   2560     // 512 rows * 80 B         (phase 1)   end 43520
#define G_OFF   0        // 32 rows * 1040 B        (phase 2)   end 33280
#define ZS_OFF  33280    // 32 * 132 floats         (phase 2)   end 50176
#define RN_OFF  50176    // 32 floats                           end 50304
#define SMEM_BYTES 50304

__global__ __launch_bounds__(256, 1) void opm_main(
    const __bf16* __restrict__ Lt, const __bf16* __restrict__ Rt,
    const __bf16* __restrict__ WoB, const float* __restrict__ mask,
    const float* __restrict__ bo, const float* __restrict__ Zraw,
    float* __restrict__ out) {
  extern __shared__ __align__(16) char smem[];
  const int jb = blockIdx.x;          // 0..11
  const int i  = blockIdx.y;          // 0..383
  const int j0 = jb * JB;
  const int tid  = threadIdx.x;
  const int wave = tid >> 5;
  const int lane = tid & 31;
  const int half = lane >> 4;
  const int ln16 = lane & 15;

  float* rnorm = (float*)(smem + RN_OFF);
  float* Zs    = (float*)(smem + ZS_OFF);

  // norm[i, j0+t] = sum_s mask[s,i]*mask[s,j]
  if (tid < JB) {
    int j = j0 + tid;
    float acc = 0.f;
    for (int s = 0; s < S_DIM; ++s)
      acc = fmaf(mask[s * R_DIM + i], mask[s * R_DIM + j], acc);
    rnorm[tid] = 1.0f / (0.001f + acc);
  }

  // ---------------- Stage A: G[c, (j,d)] = sum_s L^T x R -------------------
  v8f acc[2][8];
#pragma unroll
  for (int a = 0; a < 2; ++a)
#pragma unroll
    for (int b = 0; b < 8; ++b)
      acc[a][b] = (v8f){0.f, 0.f, 0.f, 0.f, 0.f, 0.f, 0.f, 0.f};

  const __bf16* Li = Lt + (size_t)i  * P_DIM * S_DIM;   // [32 c][256 s]
  const __bf16* Rj = Rt + (size_t)j0 * P_DIM * S_DIM;   // [1024 n][256 s]

  for (int kc = 0; kc < 8; ++kc) {          // K chunks of 32 (s dimension)
#pragma unroll
    for (int nh = 0; nh < 2; ++nh) {        // N halves of 512
      __syncthreads();
      // async stage R half: rows n in [nh*512, nh*512+512), 32 k (64 B) each
#pragma unroll
      for (int q = 0; q < 2; ++q) {
        int nl = tid + q * 256;                       // 0..511
        uint32_t goff = (uint32_t)((nh * 512 + nl) * (S_DIM * 2) + kc * 64);
        uint32_t loff = lds_off32(smem + R_OFF + nl * 80);
#pragma unroll
        for (int u = 0; u < 4; ++u)
          async_b128(loff + u * 16, goff + u * 16, Rj);
      }
      if (nh == 0 && tid < 128) {           // async stage A-chunk: 32c x 32k
        int c = tid >> 2, k = (tid & 3) * 8;
        async_b128(lds_off32(smem + A_OFF + c * 80 + k * 2),
                   (uint32_t)(c * (S_DIM * 2) + kc * 64 + k * 2), Li);
      }
      wait_async0();
      __syncthreads();

      // A fragments (16-bit A 16x32 layout: e0..7 -> K=8h+e, e8..15 -> 16+8h+e)
      FragB16 af[2];
#pragma unroll
      for (int mt = 0; mt < 2; ++mt) {
        const char* base = smem + A_OFF + (mt * 16 + ln16) * 80 + half * 16;
        af[mt].q[0] = *(const uint4*)(base);
        af[mt].q[1] = *(const uint4*)(base + 32);
      }
#pragma unroll
      for (int nt = 0; nt < 4; ++nt) {
        int idx = nh * 4 + nt;              // tile slot 0..7
        int ntg = idx * 8 + wave;           // global n-tile 0..63
        int nlr = (ntg * 16 + ln16) & 511;  // row within staged half
        const char* bp = smem + R_OFF + nlr * 80 + half * 16;
        FragB16 bf;
        bf.q[0] = *(const uint4*)(bp);
        bf.q[1] = *(const uint4*)(bp + 32);
#pragma unroll
        for (int mt = 0; mt < 2; ++mt)
          acc[mt][idx] = __builtin_amdgcn_wmma_f32_16x16x32_bf16(
              false, af[mt].v, false, bf.v, (short)0, acc[mt][idx], false, false);
      }
    }
  }
  __syncthreads();

  // ---------------- Stage B: Z[o, j] = Wo x G, K split in two halves -------
  v8f zacc[2];
  zacc[0] = (v8f){0.f, 0.f, 0.f, 0.f, 0.f, 0.f, 0.f, 0.f};
  zacc[1] = zacc[0];
  const __bf16* wrow = WoB + (size_t)(wave * 16 + ln16) * K2;   // A row = o

#pragma unroll
  for (int p = 0; p < 2; ++p) {             // K half: k in [p*512, p*512+512)
#pragma unroll
    for (int idx = 0; idx < 8; ++idx) {     // write G half (c in [p*16,p*16+16))
      int ntg = idx * 8 + wave;
      int n   = ntg * 16 + ln16;            // (j_local, d)
      int jl  = n >> 5;
      int d   = n & 31;
#pragma unroll
      for (int r = 0; r < 8; ++r) {
        int kp = (half * 8 + r) * 32 + d;   // local k = (c%16)*32 + d
        *(__bf16*)(smem + G_OFF + jl * 1040 + kp * 2) = (__bf16)acc[p][idx][r];
      }
    }
    __syncthreads();
    for (int kk = 0; kk < 16; ++kk) {       // GEMM over this K half
      int kc2 = p * 16 + kk;
      FragB16 af;                           // A row o, K = kc2*32 + {8h..}
      const __bf16* ap = wrow + kc2 * 32 + half * 8;
      af.q[0] = *(const uint4*)(ap);
      af.q[1] = *(const uint4*)(ap + 16);
#pragma unroll
      for (int nt = 0; nt < 2; ++nt) {
        int j = nt * 16 + ln16;
        const char* bp = smem + G_OFF + j * 1040 + (kk * 32 + half * 8) * 2;
        FragB16 bf;
        bf.q[0] = *(const uint4*)(bp);
        bf.q[1] = *(const uint4*)(bp + 32);
        zacc[nt] = __builtin_amdgcn_wmma_f32_16x16x32_bf16(
            false, af.v, false, bf.v, (short)0, zacc[nt], false, false);
      }
    }
    __syncthreads();
  }

  // ---------------- Epilogue ----------------------------------------------
#pragma unroll
  for (int nt = 0; nt < 2; ++nt) {
    int j = nt * 16 + ln16;
#pragma unroll
    for (int r = 0; r < 8; ++r) {
      int o = wave * 16 + half * 8 + r;
      Zs[j * 132 + o] = zacc[nt][r];
    }
  }
  __syncthreads();

  {
    int j  = tid >> 3;             // 0..31
    int o0 = (tid & 7) * 16;       // 0..112
    float rn = rnorm[j];
    size_t base = ((size_t)i * R_DIM + (j0 + j)) * O_DIM + o0;
#pragma unroll
    for (int u = 0; u < 16; u += 4) {
      float4 zr = *(const float4*)(Zraw + base + u);
      float4 bv = *(const float4*)(bo + o0 + u);
      float4 zv;
      zv.x = fmaf(Zs[j * 132 + o0 + u + 0] + bv.x, rn, zr.x);
      zv.y = fmaf(Zs[j * 132 + o0 + u + 1] + bv.y, rn, zr.y);
      zv.z = fmaf(Zs[j * 132 + o0 + u + 2] + bv.z, rn, zr.z);
      zv.w = fmaf(Zs[j * 132 + o0 + u + 3] + bv.w, rn, zr.w);
      *(float4*)(out + base + u) = zv;
    }
  }
}

// ------------------------------------------------------------------- launch
extern "C" void kernel_launch(void* const* d_in, const int* in_sizes, int n_in,
                              void* d_out, int out_size, void* d_ws, size_t ws_size,
                              hipStream_t stream) {
  const float* M     = (const float*)d_in[0];
  const float* Mmask = (const float*)d_in[1];
  const float* Zraw  = (const float*)d_in[2];
  const float* gamma = (const float*)d_in[3];
  const float* beta  = (const float*)d_in[4];
  const float* Wa    = (const float*)d_in[5];
  const float* ba    = (const float*)d_in[6];
  const float* Wb    = (const float*)d_in[7];
  const float* bb    = (const float*)d_in[8];
  const float* Wo    = (const float*)d_in[9];
  const float* bo    = (const float*)d_in[10];

  __bf16* Lt  = (__bf16*)d_ws;                       // [12288][256]
  __bf16* Rt  = Lt + (size_t)R_DIM * P_DIM * S_DIM;  // [12288][256]
  __bf16* WoB = Rt + (size_t)R_DIM * P_DIM * S_DIM;  // [128][1024]
  float* out  = (float*)d_out;

  prep_kernel<<<dim3(R_DIM / 16, S_DIM / 16), dim3(256), 0, stream>>>(
      M, Mmask, gamma, beta, Wa, ba, Wb, bb, Lt, Rt);
  wo_convert<<<dim3(512), dim3(256), 0, stream>>>(Wo, WoB);
  opm_main<<<dim3(R_DIM / JB, R_DIM), dim3(256), SMEM_BYTES, stream>>>(
      Lt, Rt, WoB, Mmask, bo, Zraw, out);
}
